// CrossCovarianceAttention_44358422233488
// MI455X (gfx1250) — matrix operand order repaired and checked by reading, not compiled, not run on hardware
//
#include <hip/hip_runtime.h>
#include <hip/hip_bf16.h>

// ---------------------------------------------------------------------------
// CrossCovarianceAttention (XCiT) for MI455X / gfx1250, wave32 + WMMA bf16.
// B=8, N=4096, DIM=768, H=8, DH=96.
// All GEMMs via v_wmma_f32_16x16x32_bf16 (fp32 accumulate), stats in fp32.
// QK^T uses a deterministic split-K (4 partials, fixed-order reduce) so the
// covariance stage fills the machine without fp-atomic nondeterminism.
// ---------------------------------------------------------------------------

typedef __bf16 bf16_t;
typedef __attribute__((ext_vector_type(16))) __bf16 v16bf;
typedef __attribute__((ext_vector_type(8)))  float  v8f;

#define DEVI __device__ __forceinline__

static constexpr int Bb    = 8;
static constexpr int Nn    = 4096;
static constexpr int DIM   = 768;
static constexpr int Hh    = 8;
static constexpr int DH    = 96;
static constexpr int ROWS  = 3 * DIM;         // 2304 fused q/k/v channels
static constexpr int XPAD  = DIM + 8;         // 776: LDS row stride (4-bank skew)
static constexpr int KPART = 4;               // split-K parts for QK^T

DEVI v8f wmma_bf16(v16bf a, v16bf b, v8f c) {
  // (neg_a, A, neg_b, B, c_mod, C, reuse_a, reuse_b)
  return __builtin_amdgcn_wmma_f32_16x16x32_bf16(false, a, false, b,
                                                 (short)0, c, false, false);
}

// A fragment: 16(M) x 32(K), source row-major [m][k], ld in elements.
// lanes 0-15: m=lane, K = 0..7 and 16..23 ; lanes 16-31: m=lane-16, K = 8..15 and 24..31
DEVI v16bf load_frag_a(const bf16_t* p, int ld, int lane) {
  const int half = lane >> 4, m = lane & 15;
  const bf16_t* r = p + (long)m * ld + half * 8;
  v16bf out;
  uint4* o = reinterpret_cast<uint4*>(&out);
  o[0] = *reinterpret_cast<const uint4*>(r);        // K = half*8 .. +7
  o[1] = *reinterpret_cast<const uint4*>(r + 16);   // K = 16+half*8 .. +7
  return out;
}

// B fragment: 32(K) x 16(N), source stored [n][k] (k contiguous), ld in elems.
// lanes 0-15: n=lane, K = 0..7 and 16..23 ; lanes 16-31: n=lane-16, K = 8..15 and 24..31
DEVI v16bf load_frag_b(const bf16_t* p, int ld, int lane) {
  const int n = lane & 15, kb = (lane >> 4) * 8;
  const bf16_t* r = p + (long)n * ld + kb;
  v16bf out;
  uint4* o = reinterpret_cast<uint4*>(&out);
  o[0] = *reinterpret_cast<const uint4*>(r);
  o[1] = *reinterpret_cast<const uint4*>(r + 16);
  return out;
}

// ---------------------------------------------------------------------------
// K0: convert weights to bf16 (fused Wqkv [2304][768], Wo [768][768]); concat
//     biases into fp32[2304].
// ---------------------------------------------------------------------------
__global__ __launch_bounds__(256)
void k0_convert(const float* __restrict__ Wq, const float* __restrict__ Wk,
                const float* __restrict__ Wv, const float* __restrict__ Wo,
                const float* __restrict__ bq, const float* __restrict__ bk,
                const float* __restrict__ bv,
                bf16_t* __restrict__ Wqkv, bf16_t* __restrict__ Wob,
                float* __restrict__ bias) {
  const int i = blockIdx.x * 256 + threadIdx.x;
  const int NW = ROWS * DIM;                      // 2304*768
  if (i < NW) {
    const int m = i / (DIM * DIM);
    const int j = i - m * (DIM * DIM);
    const float s = (m == 0) ? Wq[j] : (m == 1) ? Wk[j] : Wv[j];
    Wqkv[i] = (bf16_t)s;
  } else if (i < NW + DIM * DIM) {
    Wob[i - NW] = (bf16_t)Wo[i - NW];
  }
  if (i < ROWS) {
    bias[i] = (i < DIM) ? bq[i] : (i < 2 * DIM) ? bk[i - DIM] : bv[i - 2 * DIM];
  }
}

// ---------------------------------------------------------------------------
// K1: fused QKV projection. QKV[b][row][n] (bf16), row = matrix*768 + h*96 + c.
// One WG = one batch b, 32 tokens. x tile staged in LDS as bf16; the compiler
// keeps all 48 B-fragments register-resident and streams weight A-fragments.
// ---------------------------------------------------------------------------
__global__ __launch_bounds__(256)
void k1_qkv_proj(const float* __restrict__ x, const bf16_t* __restrict__ Wqkv,
                 const float* __restrict__ bias, bf16_t* __restrict__ QKV) {
  const int b = blockIdx.y;
  const int nbase = blockIdx.x * 32;
  const int tid = threadIdx.x;
  __shared__ __attribute__((aligned(16))) bf16_t xs[32 * XPAD];

  // stage x tile: 32 tokens x 768, fp32 -> bf16
  for (int t = tid; t < 32 * DIM / 4; t += 256) {
    const int tok = t / (DIM / 4);
    const int d = (t - tok * (DIM / 4)) * 4;
    const float4 f = *reinterpret_cast<const float4*>(
        x + ((long)b * Nn + nbase + tok) * DIM + d);
    bf16_t* dst = xs + tok * XPAD + d;
    dst[0] = (bf16_t)f.x; dst[1] = (bf16_t)f.y;
    dst[2] = (bf16_t)f.z; dst[3] = (bf16_t)f.w;
  }
  __syncthreads();

  const int wave = tid >> 5, lane = tid & 31;
  const int halfl = lane >> 4, nl = lane & 15;

  for (int rt = wave; rt < ROWS / 16; rt += 8) {       // 144 row tiles
    const int row0 = rt * 16;
    const bf16_t* wrow = Wqkv + (long)row0 * DIM;
    __builtin_prefetch(wrow, 0, 3);                    // near-scope prefetch
    v8f acc0 = {}, acc1 = {};
    for (int ks = 0; ks < DIM / 32; ++ks) {            // 24 k-steps
      const v16bf a  = load_frag_a(wrow + ks * 32, DIM, lane);
      const v16bf b0 = load_frag_b(xs + ks * 32, XPAD, lane);             // tokens 0-15
      const v16bf b1 = load_frag_b(xs + 16 * XPAD + ks * 32, XPAD, lane); // 16-31
      acc0 = wmma_bf16(a, b0, acc0);
      acc1 = wmma_bf16(a, b1, acc1);
    }
    for (int r = 0; r < 8; ++r) {
      const int row = row0 + r + 8 * halfl;
      const float bv_ = bias[row];
      bf16_t* dst = QKV + ((long)b * ROWS + row) * Nn + nbase;
      dst[nl]      = (bf16_t)(acc0[r] + bv_);
      dst[16 + nl] = (bf16_t)(acc1[r] + bv_);
    }
  }
}

// ---------------------------------------------------------------------------
// K2: per-row (q,k only) sum of squares over N -> rnorm = 1/max(sqrt,eps).
// One wave per row; 8 waves per WG.
// ---------------------------------------------------------------------------
__global__ __launch_bounds__(256)
void k2_rnorm(const bf16_t* __restrict__ QKV, float* __restrict__ rnorm) {
  const int wave = threadIdx.x >> 5, lane = threadIdx.x & 31;
  const int rg = blockIdx.x * 8 + wave;          // 0 .. B*1536-1
  const int b = rg / (2 * DIM);
  const int row = rg - b * (2 * DIM);            // q rows 0..767, k rows 768..1535
  const bf16_t* src = QKV + ((long)b * ROWS + row) * Nn;

  float s = 0.f;
  for (int i = lane * 4; i < Nn; i += 128) {
    union { uint2 u; bf16_t h[4]; } cv;
    cv.u = *reinterpret_cast<const uint2*>(src + i);
    for (int j = 0; j < 4; ++j) { const float v = (float)cv.h[j]; s += v * v; }
  }
  for (int off = 16; off; off >>= 1) s += __shfl_xor(s, off);
  if (lane == 0) rnorm[b * (2 * DIM) + row] = 1.0f / fmaxf(sqrtf(s), 1e-12f);
}

// ---------------------------------------------------------------------------
// K3a: raw S partials = Q K^T over an N/4 slice per WG (bf16 WMMA).
// Grid (KPART, B*H): 256 WGs. Partials stored fp32, fixed layout.
// ---------------------------------------------------------------------------
__global__ __launch_bounds__(256)
void k3a_qk_part(const bf16_t* __restrict__ QKV, float* __restrict__ Spart) {
  const int part = blockIdx.x;                   // 0..3
  const int bh = blockIdx.y, b = bh >> 3, h = bh & 7;
  const int n0 = part * (Nn / KPART);
  const bf16_t* Qp = QKV + ((long)b * ROWS + h * DH) * Nn + n0;
  const bf16_t* Kp = QKV + ((long)b * ROWS + DIM + h * DH) * Nn + n0;

  const int wave = threadIdx.x >> 5, lane = threadIdx.x & 31;
  const int halfl = lane >> 4, nl = lane & 15;

  for (int t = wave; t < 36; t += 8) {           // 6x6 tile grid of 96x96
    const int ct = t / 6, et = t - 6 * (t / 6);
    v8f acc = {};
    for (int ks = 0; ks < (Nn / KPART) / 32; ++ks) {   // 32 k-steps
      const v16bf a = load_frag_a(Qp + (long)ct * 16 * Nn + ks * 32, Nn, lane);
      const v16bf v = load_frag_b(Kp + (long)et * 16 * Nn + ks * 32, Nn, lane);
      acc = wmma_bf16(a, v, acc);
    }
    const int e = et * 16 + nl;
    float* dst = Spart + (((long)bh * KPART + part) * DH) * DH;
    for (int r = 0; r < 8; ++r) {
      const int c = ct * 16 + r + 8 * halfl;
      dst[c * DH + e] = acc[r];
    }
  }
}

// ---------------------------------------------------------------------------
// K3b: fixed-order reduce of the 4 partials, scale by rq[c]*rk[e]*temp[h],
// fp32 softmax over e, store attn bf16 [96][96]. One WG per (b,h).
// ---------------------------------------------------------------------------
__global__ __launch_bounds__(256)
void k3b_softmax(const float* __restrict__ Spart, const float* __restrict__ rnorm,
                 const float* __restrict__ temp, bf16_t* __restrict__ attn) {
  const int bh = blockIdx.x, b = bh >> 3, h = bh & 7;
  __shared__ __attribute__((aligned(16))) float S[DH * 100];
  const int tid = threadIdx.x;
  const float th = temp[h];

  const float* base = Spart + ((long)bh * KPART) * DH * DH;
  for (int t = tid; t < DH * DH; t += 256) {
    const int c = t / DH, e = t - c * DH;
    float s = 0.f;
    for (int p = 0; p < KPART; ++p) s += base[(long)p * DH * DH + t];  // fixed order
    const float rq = rnorm[b * (2 * DIM) + h * DH + c];
    const float rk = rnorm[b * (2 * DIM) + DIM + h * DH + e];
    S[c * 100 + e] = s * rq * rk * th;
  }
  __syncthreads();

  const int wave = tid >> 5, lane = tid & 31;
  // softmax: each wave owns 12 rows of 96
  for (int rr = 0; rr < 12; ++rr) {
    const int c = wave * 12 + rr;
    const float v0 = S[c * 100 + lane];
    const float v1 = S[c * 100 + 32 + lane];
    const float v2 = S[c * 100 + 64 + lane];
    float m = fmaxf(v0, fmaxf(v1, v2));
    for (int off = 16; off; off >>= 1) m = fmaxf(m, __shfl_xor(m, off));
    const float e0 = __expf(v0 - m), e1 = __expf(v1 - m), e2 = __expf(v2 - m);
    float s = e0 + e1 + e2;
    for (int off = 16; off; off >>= 1) s += __shfl_xor(s, off);
    const float inv = 1.0f / s;
    bf16_t* dst = attn + ((long)bh * DH + c) * DH;
    dst[lane]      = (bf16_t)(e0 * inv);
    dst[lane + 32] = (bf16_t)(e1 * inv);
    dst[lane + 64] = (bf16_t)(e2 * inv);
  }
}

// ---------------------------------------------------------------------------
// K4: ctx = attn @ V per (b,h), 128-token tiles. attn and transposed V staged
//     in LDS; result written bf16 packed (b128) into [b][n][h*96+c] layout.
// ---------------------------------------------------------------------------
__global__ __launch_bounds__(256)
void k4_av(const bf16_t* __restrict__ QKV, const bf16_t* __restrict__ attn,
           bf16_t* __restrict__ ctx) {
  const int bh = blockIdx.y, b = bh >> 3, h = bh & 7;
  const int nbase = blockIdx.x * 128;
  const int tid = threadIdx.x;
  const bf16_t* Vp = QKV + ((long)b * ROWS + 2 * DIM + h * DH) * Nn;

  __shared__ __attribute__((aligned(16))) bf16_t At[DH * 104];
  __shared__ __attribute__((aligned(16))) bf16_t Vt[128 * 104];

  for (int t = tid; t < DH * DH; t += 256) {            // attn [c][e]
    const int c = t / DH, e = t - c * DH;
    At[c * 104 + e] = attn[(long)bh * DH * DH + t];
  }
  for (int t = tid; t < DH * 128; t += 256) {           // V^T -> [n][e]
    const int e = t / 128, j = t - e * 128;
    Vt[j * 104 + e] = Vp[(long)e * Nn + nbase + j];
  }
  __syncthreads();

  const int wave = tid >> 5, lane = tid & 31;
  const int halfl = lane >> 4, nl = lane & 15;

  for (int t = wave; t < 48; t += 8) {                  // 6 c-tiles x 8 n-tiles
    const int ct = t % 6, jt = t / 6;
    v8f acc = {};
    for (int ks = 0; ks < 3; ++ks) {                    // K = 96 = 3*32
      const v16bf a = load_frag_a(At + ct * 16 * 104 + ks * 32, 104, lane);
      const v16bf v = load_frag_b(Vt + jt * 16 * 104 + ks * 32, 104, lane);
      acc = wmma_bf16(a, v, acc);
    }
    const int n = nbase + jt * 16 + nl;
    const int chbase = h * DH + ct * 16 + 8 * halfl;    // 8 consecutive channels
    bf16_t pk[8];
    for (int r = 0; r < 8; ++r) pk[r] = (bf16_t)acc[r];
    *reinterpret_cast<uint4*>(ctx + ((long)b * Nn + n) * DIM + chbase) =
        *reinterpret_cast<uint4*>(pk);
  }
}

// ---------------------------------------------------------------------------
// K5: out = ctx @ Wo^T + bo (fp32 out). ctx tile staged in LDS.
// ---------------------------------------------------------------------------
__global__ __launch_bounds__(256)
void k5_out_proj(const bf16_t* __restrict__ ctx, const bf16_t* __restrict__ Wob,
                 const float* __restrict__ bo, float* __restrict__ out) {
  const int b = blockIdx.y;
  const int nbase = blockIdx.x * 32;
  const int tid = threadIdx.x;
  __shared__ __attribute__((aligned(16))) bf16_t qs[32 * XPAD];

  for (int t = tid; t < 32 * DIM / 8; t += 256) {       // b128 copy-in
    const int tok = t / (DIM / 8);
    const int d = (t - tok * (DIM / 8)) * 8;
    *reinterpret_cast<uint4*>(qs + tok * XPAD + d) =
        *reinterpret_cast<const uint4*>(ctx + ((long)b * Nn + nbase + tok) * DIM + d);
  }
  __syncthreads();

  const int wave = tid >> 5, lane = tid & 31;
  const int halfl = lane >> 4, nl = lane & 15;

  for (int t = wave; t < 96; t += 8) {                  // 2 tok-tiles x 48 o-tiles
    const int ot = t >> 1, tt = t & 1;
    v8f acc = {};
    for (int ks = 0; ks < DIM / 32; ++ks) {             // 24 k-steps
      const v16bf a = load_frag_a(qs + tt * 16 * XPAD + ks * 32, XPAD, lane);
      const v16bf v = load_frag_b(Wob + (long)ot * 16 * DIM + ks * 32, DIM, lane);
      acc = wmma_bf16(a, v, acc);
    }
    const int o = ot * 16 + nl;
    const float bov = bo[o];
    for (int r = 0; r < 8; ++r) {
      const int tok = nbase + tt * 16 + r + 8 * halfl;
      out[((long)b * Nn + tok) * DIM + o] = acc[r] + bov;
    }
  }
}

// ---------------------------------------------------------------------------
// Host-side launcher. Inputs: x, Wq, bq, Wk, bk, Wv, bv, temp, Wo, bo (fp32).
// ---------------------------------------------------------------------------
extern "C" void kernel_launch(void* const* d_in, const int* in_sizes, int n_in,
                              void* d_out, int out_size, void* d_ws, size_t ws_size,
                              hipStream_t stream) {
  const float* x    = (const float*)d_in[0];
  const float* Wq   = (const float*)d_in[1];
  const float* bq   = (const float*)d_in[2];
  const float* Wk   = (const float*)d_in[3];
  const float* bk   = (const float*)d_in[4];
  const float* Wv   = (const float*)d_in[5];
  const float* bv   = (const float*)d_in[6];
  const float* temp = (const float*)d_in[7];
  const float* Wo   = (const float*)d_in[8];
  const float* bo   = (const float*)d_in[9];
  float* out = (float*)d_out;

  // workspace layout (bytes)
  char* w = (char*)d_ws;
  auto align256 = [](size_t v) { return (v + 255) & ~(size_t)255; };
  size_t off = 0;
  bf16_t* Wqkv = (bf16_t*)(w + off); off = align256(off + (size_t)ROWS * DIM * 2);
  bf16_t* Wob  = (bf16_t*)(w + off); off = align256(off + (size_t)DIM * DIM * 2);
  float*  bias = (float*)(w + off);  off = align256(off + (size_t)ROWS * 4);
  bf16_t* QKV  = (bf16_t*)(w + off); off = align256(off + (size_t)Bb * ROWS * Nn * 2);
  float*  rnorm= (float*)(w + off);  off = align256(off + (size_t)Bb * 2 * DIM * 4);
  float*  Spart= (float*)(w + off);  off = align256(off + (size_t)Bb * Hh * KPART * DH * DH * 4);
  bf16_t* attn = (bf16_t*)(w + off); off = align256(off + (size_t)Bb * Hh * DH * DH * 2);
  bf16_t* ctx  = (bf16_t*)(w + off); off = align256(off + (size_t)Bb * Nn * DIM * 2);
  (void)ws_size; (void)n_in; (void)in_sizes; (void)out_size;

  k0_convert<<<(ROWS * DIM + DIM * DIM + 255) / 256, 256, 0, stream>>>(
      Wq, Wk, Wv, Wo, bq, bk, bv, Wqkv, Wob, bias);

  k1_qkv_proj<<<dim3(Nn / 32, Bb), 256, 0, stream>>>(x, Wqkv, bias, QKV);

  k2_rnorm<<<(Bb * 2 * DIM) / 8, 256, 0, stream>>>(QKV, rnorm);

  k3a_qk_part<<<dim3(KPART, Bb * Hh), 256, 0, stream>>>(QKV, Spart);

  k3b_softmax<<<Bb * Hh, 256, 0, stream>>>(Spart, rnorm, temp, attn);

  k4_av<<<dim3(Nn / 128, Bb * Hh), 256, 0, stream>>>(QKV, attn, ctx);

  k5_out_proj<<<dim3(Nn / 32, Bb), 256, 0, stream>>>(ctx, Wob, bo, out);
}
